// GroupGemmOp_32349693674007
// MI455X (gfx1250) — matrix operand order repaired and checked by reading, not compiled, not run on hardware
//
#include <hip/hip_runtime.h>
#include <hip/hip_bf16.h>

typedef __attribute__((ext_vector_type(16))) __bf16 v16bf;
typedef __attribute__((ext_vector_type(8)))  __bf16 v8bf;
typedef __attribute__((ext_vector_type(4)))  __bf16 v4bf;
typedef __attribute__((ext_vector_type(8)))  float  v8f;
typedef __attribute__((ext_vector_type(4)))  float  v4f;

namespace {
constexpr int BM   = 128;   // block tile M
constexpr int BN   = 256;   // block tile N (4 N-blocks across NDIM)
constexpr int KT   = 32;    // K step (bf16 WMMA K)
constexpr int LDK  = 40;    // padded LDS row (elements): 80B rows, 16B-aligned, bank-friendly
constexpr int KDIM = 1024;
constexpr int NDIM = 1024;
constexpr int NEXP = 8;
}

#define WMMA_BF16(a, b, c) \
    __builtin_amdgcn_wmma_f32_16x16x32_bf16(false, (a), false, (b), (short)0, (c), false, false)

__device__ __forceinline__ v16bf cat16(v8bf lo, v8bf hi) {
    v16bf r;
#pragma unroll
    for (int i = 0; i < 8; ++i) { r[i] = lo[i]; r[i + 8] = hi[i]; }
    return r;
}

struct GGParams {
    const float* A[NEXP];
    const float* B[NEXP];
    long long    coff[NEXP];     // output offset (elements) per expert
    int          bstart[NEXP + 1]; // flat tile-index range per expert
};

// Fused grouped GEMM: one launch covers all experts. Per block: C tile 128x256,
// 8 waves each owning 64x64 (4x4 WMMA tiles), fp32 via 3-term bf16 hi/lo split.
__global__ __launch_bounds__(256)
void grouped_gemm_bf16x3_wmma(GGParams p, float* __restrict__ out) {
    // ---- expert lookup (scalar) ----
    const int bid = blockIdx.x;
    int e = 0;
#pragma unroll
    for (int i = 1; i < NEXP; ++i)
        if (bid >= p.bstart[i]) e = i;
    const int local = bid - p.bstart[e];
    const int mBlock = (local >> 2) * BM;     // NDIM/BN == 4 tiles in N
    const int nBlock = (local & 3) * BN;

    const float* __restrict__ A = p.A[e];
    const float* __restrict__ B = p.B[e];
    float* __restrict__ C = out + p.coff[e];

    // ---- LDS: double-buffered bf16 hi/lo tiles; A [M][K], B transposed [N][K] ----
    __shared__ __bf16 Ah[2][BM * LDK];
    __shared__ __bf16 Al[2][BM * LDK];
    __shared__ __bf16 Bh[2][BN * LDK];
    __shared__ __bf16 Bl[2][BN * LDK];

    const int tid  = threadIdx.x;
    const int lane = tid & 31;
    const int half = lane >> 4;
    const int l16  = lane & 15;
    const int wid  = tid >> 5;
    const int wm   = wid >> 2;    // 2 waves along M (64 rows)
    const int wn   = wid & 3;     // 4 waves along N (64 cols)

    // ---- per-thread global pointers, advanced by constant strides ----
    const float* aPtr[4];
    const float* bPtr[8];
    int aOff[4];          // LDS element offsets for A staging
    int bOff[8];          // LDS element offsets (base) for B staging (transposed)
#pragma unroll
    for (int i = 0; i < 4; ++i) {
        const int f  = tid + 256 * i;
        const int ar = f >> 3;                // 0..127
        const int ac = (f & 7) << 2;          // 0..28
        aPtr[i] = A + (size_t)(mBlock + ar) * KDIM + ac;
        aOff[i] = ar * LDK + ac;
    }
#pragma unroll
    for (int i = 0; i < 8; ++i) {
        const int f  = tid + 256 * i;
        const int br = f >> 6;                // K row 0..31
        const int bc = (f & 63) << 2;         // N col 0..252
        bPtr[i] = B + (size_t)br * NDIM + nBlock + bc;
        bOff[i] = bc * LDK + br;              // +j*LDK per element j
    }

    v8f acc[4][4];
#pragma unroll
    for (int i = 0; i < 4; ++i)
#pragma unroll
        for (int j = 0; j < 4; ++j) {
            v8f z = {0.f, 0.f, 0.f, 0.f, 0.f, 0.f, 0.f, 0.f};
            acc[i][j] = z;
        }

    v4f aReg[4], bReg[8];

    auto loadGlobal = [&]() {
#pragma unroll
        for (int i = 0; i < 4; ++i) { aReg[i] = *(const v4f*)aPtr[i]; aPtr[i] += KT; }
#pragma unroll
        for (int i = 0; i < 8; ++i) { bReg[i] = *(const v4f*)bPtr[i]; bPtr[i] += (size_t)KT * NDIM; }
    };

    auto storeLds = [&](int buf) {
#pragma unroll
        for (int i = 0; i < 4; ++i) {
            v4bf h4, l4;
#pragma unroll
            for (int j = 0; j < 4; ++j) {
                const float x = aReg[i][j];
                const __bf16 h = (__bf16)x;
                h4[j] = h;
                l4[j] = (__bf16)(x - (float)h);
            }
            *(v4bf*)&Ah[buf][aOff[i]] = h4;
            *(v4bf*)&Al[buf][aOff[i]] = l4;
        }
#pragma unroll
        for (int i = 0; i < 8; ++i) {
#pragma unroll
            for (int j = 0; j < 4; ++j) {
                const float x = bReg[i][j];
                const __bf16 h = (__bf16)x;
                Bh[buf][bOff[i] + j * LDK] = h;
                Bl[buf][bOff[i] + j * LDK] = (__bf16)(x - (float)h);
            }
        }
    };

    // A fragment (16x32 bf16, ISA 7.12.2): lanes 0-15 -> K [0,8)+[16,24),
    // lanes 16-31 -> K [8,16)+[24,32), row M = l16.
    auto loadA = [&](const __bf16* base, int mt) -> v16bf {
        const __bf16* q = base + (size_t)(wm * 64 + mt * 16 + l16) * LDK + half * 8;
        return cat16(*(const v8bf*)q, *(const v8bf*)(q + 16));
    };
    // B fragment (32x16 bf16): lane = column N, lanes 0-15 -> K [0,8)+[16,24), lanes 16-31 -> K [8,16)+[24,32).
    auto loadB = [&](const __bf16* base, int nt) -> v16bf {
        const __bf16* q = base + (size_t)(wn * 64 + nt * 16 + l16) * LDK + half * 8;
        return cat16(*(const v8bf*)q, *(const v8bf*)(q + 16));
    };

    constexpr int NS = KDIM / KT;  // 32 K-steps

    loadGlobal();
    storeLds(0);
    __syncthreads();

    for (int kt = 0; kt < NS; ++kt) {
        const int cur = kt & 1;
        if (kt + 1 < NS) loadGlobal();          // prefetch next K tile into regs

        // Cache all A fragments for this K-step (8 x v16bf = 64 VGPRs).
        v16bf ahf[4], alf[4];
#pragma unroll
        for (int mt = 0; mt < 4; ++mt) {
            ahf[mt] = loadA(Ah[cur], mt);
            alf[mt] = loadA(Al[cur], mt);
        }

#pragma unroll
        for (int nt = 0; nt < 4; ++nt) {
            const v16bf bh = loadB(Bh[cur], nt);
            const v16bf bl = loadB(Bl[cur], nt);
#pragma unroll
            for (int mt = 0; mt < 4; ++mt) {
                acc[mt][nt] = WMMA_BF16(ahf[mt], bh, acc[mt][nt]);  // hi*hi
                acc[mt][nt] = WMMA_BF16(ahf[mt], bl, acc[mt][nt]);  // hi*lo
                acc[mt][nt] = WMMA_BF16(alf[mt], bh, acc[mt][nt]);  // lo*hi
            }
        }

        if (kt + 1 < NS) storeLds((kt + 1) & 1);
        __syncthreads();
    }

    // Epilogue: C/D layout — VGPR v: M = v + 8*half; N = l16.
    float* Cb = C + (size_t)mBlock * NDIM + nBlock;
    const int ccolBase = wn * 64 + l16;
#pragma unroll
    for (int mt = 0; mt < 4; ++mt) {
#pragma unroll
        for (int v = 0; v < 8; ++v) {
            const int r = wm * 64 + mt * 16 + half * 8 + v;
            float* rowp = Cb + (size_t)r * NDIM + ccolBase;
#pragma unroll
            for (int nt = 0; nt < 4; ++nt)
                rowp[nt * 16] = acc[mt][nt][v];
        }
    }
}

extern "C" void kernel_launch(void* const* d_in, const int* in_sizes, int n_in,
                              void* d_out, int out_size, void* d_ws, size_t ws_size,
                              hipStream_t stream) {
    (void)d_ws; (void)ws_size; (void)out_size;
    GGParams p;
    long long off = 0;
    int bacc = 0;
    const int n_experts = n_in / 2;
    for (int e = 0; e < NEXP; ++e) {
        const int ei = (e < n_experts) ? e : (n_experts - 1);
        p.A[e] = (const float*)d_in[2 * ei];
        p.B[e] = (const float*)d_in[2 * ei + 1];
        if (e < n_experts) {
            const int M = in_sizes[2 * e] / KDIM;
            p.coff[e]   = off;
            p.bstart[e] = bacc;
            off  += (long long)M * NDIM;
            bacc += (M / BM) * (NDIM / BN);
        } else {
            p.coff[e]   = off;
            p.bstart[e] = bacc;
        }
    }
    p.bstart[NEXP] = bacc;
    grouped_gemm_bf16x3_wmma<<<dim3(bacc), dim3(256), 0, stream>>>(p, (float*)d_out);
}
